// FractalLinear_71717363908756
// MI455X (gfx1250) — compile-verified, matrix operand order checked by reading, not run on hardware
//
#include <hip/hip_runtime.h>

// ---------------------------------------------------------------------------
// FractalLinear on MI455X (gfx1250):
//   1) transcribe: 20x  w = a*w + b*sin(w)   (f32, elementwise) -> bf16 weight
//   2) convert x f32 -> bf16
//   3) GEMM out[t,o] = sum_i x[t,i]*w[o,i] + bias[o]  via v_wmma_f32_16x16x32_bf16
//      with double-buffered async global->LDS staging (ASYNCcnt pipeline).
// ---------------------------------------------------------------------------

typedef __attribute__((ext_vector_type(16))) __bf16 v16bf;
typedef __attribute__((ext_vector_type(8)))  __bf16 v8bf;
typedef __attribute__((ext_vector_type(4)))  __bf16 v4bf;
typedef __attribute__((ext_vector_type(8)))  float  v8f;
typedef __attribute__((ext_vector_type(4)))  int    v4i;

#define TOKENS 2048
#define INF    4096
#define OUTF   4096
#define NITER  20

// ---- async global->LDS support (guarded: falls back to sync copy) ----------
#if defined(__has_builtin)
# if __has_builtin(__builtin_amdgcn_global_load_async_to_lds_b128)
#  define HAVE_ASYNC_LDS 1
# endif
# if __has_builtin(__builtin_amdgcn_s_wait_asynccnt)
#  define HAVE_WAIT_ASYNC_BUILTIN 1
# endif
#endif

// Builtin signature (from hipcc diagnostic): arg0 = int4* in AS(1) (global),
// arg1 = int4* in AS(3) (LDS), arg2 = imm offset, arg3 = imm cpol.
typedef __attribute__((address_space(1))) v4i gv4i_t;
typedef __attribute__((address_space(3))) v4i lv4i_t;

__device__ __forceinline__ void copy16_to_lds(void* lds_dst, const void* gsrc) {
#if defined(HAVE_ASYNC_LDS)
  // AS(3) is a 32-bit offset space; flat-addr low 32 bits are the LDS offset
  // (ISA 10.2). Integer-cast route avoids generic->AS cast legality issues.
  gv4i_t* g = (gv4i_t*)(uintptr_t)gsrc;
  lv4i_t* l = (lv4i_t*)(unsigned int)(uintptr_t)lds_dst;
  __builtin_amdgcn_global_load_async_to_lds_b128(g, l, /*offset=*/0, /*cpol=*/0);
#else
  *(uint4*)lds_dst = *(const uint4*)gsrc;
#endif
}

__device__ __forceinline__ void wait_async_copies() {
#if defined(HAVE_ASYNC_LDS)
# if defined(HAVE_WAIT_ASYNC_BUILTIN)
  __builtin_amdgcn_s_wait_asynccnt(0);
# else
  asm volatile("s_wait_asynccnt 0x0" ::: "memory");
# endif
#endif
}

// ---------------- Stage 1: fractal DNA transcription (f32 -> bf16) ----------
__global__ __launch_bounds__(256) void fractal_transcribe_bf16(
    const float* __restrict__ seed, const float* __restrict__ da,
    const float* __restrict__ db, __bf16* __restrict__ wbf) {
  const size_t idx = ((size_t)blockIdx.x * blockDim.x + threadIdx.x) * 4;
  float4 w = *(const float4*)(seed + idx);
#pragma unroll
  for (int it = 0; it < NITER; ++it) {
    const float a = da[it];   // uniform -> scalar loads
    const float b = db[it];
    w.x = fmaf(a, w.x, b * sinf(w.x));
    w.y = fmaf(a, w.y, b * sinf(w.y));
    w.z = fmaf(a, w.z, b * sinf(w.z));
    w.w = fmaf(a, w.w, b * sinf(w.w));
  }
  v4bf o;
  o[0] = (__bf16)w.x; o[1] = (__bf16)w.y; o[2] = (__bf16)w.z; o[3] = (__bf16)w.w;
  *(v4bf*)(wbf + idx) = o;
}

// ---------------- Stage 2: x f32 -> bf16 ------------------------------------
__global__ __launch_bounds__(256) void cvt_f32_to_bf16(
    const float* __restrict__ in, __bf16* __restrict__ out) {
  const size_t idx = ((size_t)blockIdx.x * blockDim.x + threadIdx.x) * 4;
  const float4 v = *(const float4*)(in + idx);
  v4bf o;
  o[0] = (__bf16)v.x; o[1] = (__bf16)v.y; o[2] = (__bf16)v.z; o[3] = (__bf16)v.w;
  *(v4bf*)(out + idx) = o;
}

// ---------------- Stage 3: WMMA GEMM ----------------------------------------
// Block: 256 threads (8 wave32), 128x128 C tile. Waves 4(M) x 2(N),
// each wave: 32x64 = 2x4 fragments of 16x16, K-step 32, double-buffered LDS.
__global__ __launch_bounds__(256) void fractal_gemm_wmma(
    const __bf16* __restrict__ A,   // x_bf   [TOKENS][INF] row-major
    const __bf16* __restrict__ B,   // w_bf   [OUTF][INF]  row-major (== B^T tiles)
    const float*  __restrict__ bias,
    float*        __restrict__ C) { // out    [TOKENS][OUTF]
  constexpr int BK   = 32;
  constexpr int LDSS = BK + 8;      // 80B row stride: 16B aligned, conflict-free
  constexpr int NT   = INF / BK;    // 128 K-tiles

  __shared__ __bf16 As[2][128][LDSS];
  __shared__ __bf16 Bs[2][128][LDSS];

  const int t    = threadIdx.x;
  const int lane = t & 31;
  const int wid  = t >> 5;          // 0..7
  const int wm0  = (wid & 3) * 32;  // wave M offset in block tile
  const int wn0  = (wid >> 2) * 64; // wave N offset in block tile
  const int lr   = lane & 15;       // fragment row/col within 16
  const int lh   = lane >> 4;       // half select (K-interleave / M-high)

  const int m0 = blockIdx.y * 128;
  const int n0 = blockIdx.x * 128;

  // global -> LDS staging: each thread moves 32B of A and 32B of B per K step
  const int row  = t >> 1;          // 0..127
  const int colh = (t & 1) * 16;    // element offset 0 or 16

  const __bf16* gA = A + (size_t)(m0 + row) * INF + colh;
  const __bf16* gB = B + (size_t)(n0 + row) * INF + colh;

  v8f acc[2][4] = {};

  // Prologue: stage tile 0 into buffer 0.
  {
    copy16_to_lds(&As[0][row][colh],     gA);
    copy16_to_lds(&As[0][row][colh + 8], gA + 8);
    copy16_to_lds(&Bs[0][row][colh],     gB);
    copy16_to_lds(&Bs[0][row][colh + 8], gB + 8);
    wait_async_copies();
    __syncthreads();
  }

  for (int kt = 0; kt < NT; ++kt) {
    const int b = kt & 1;

    // Issue async staging of the next K-slice into the other buffer while
    // this buffer feeds the matrix pipe.
    if (kt + 1 < NT) {
      const __bf16* pA = gA + (size_t)(kt + 1) * BK;
      const __bf16* pB = gB + (size_t)(kt + 1) * BK;
      copy16_to_lds(&As[b ^ 1][row][colh],     pA);
      copy16_to_lds(&As[b ^ 1][row][colh + 8], pA + 8);
      copy16_to_lds(&Bs[b ^ 1][row][colh],     pB);
      copy16_to_lds(&Bs[b ^ 1][row][colh + 8], pB + 8);
    }

    // A fragment: lanes 0-15 -> K 0-7 & 16-23 of row (wm0+i*16+lr),
    //             lanes 16-31 -> K 8-15 & 24-31. Two ds_read_b128 each.
    v16bf af[2];
#pragma unroll
    for (int i = 0; i < 2; ++i) {
      const __bf16* p = &As[b][wm0 + i * 16 + lr][lh * 8];
      v8bf lo = *(const v8bf*)(p);
      v8bf hi = *(const v8bf*)(p + 16);
      af[i] = __builtin_shufflevector(lo, hi, 0,1,2,3,4,5,6,7,8,9,10,11,12,13,14,15);
    }
    // B fragment: identical pattern, lane = output-feature column.
    v16bf bfr[4];
#pragma unroll
    for (int j = 0; j < 4; ++j) {
      const __bf16* p = &Bs[b][wn0 + j * 16 + lr][lh * 8];
      v8bf lo = *(const v8bf*)(p);
      v8bf hi = *(const v8bf*)(p + 16);
      bfr[j] = __builtin_shufflevector(lo, hi, 0,1,2,3,4,5,6,7,8,9,10,11,12,13,14,15);
    }

#pragma unroll
    for (int i = 0; i < 2; ++i)
#pragma unroll
      for (int j = 0; j < 4; ++j)
        acc[i][j] = __builtin_amdgcn_wmma_f32_16x16x32_bf16(
            /*neg_a=*/false, af[i], /*neg_b=*/false, bfr[j],
            /*c_mod=*/(short)0, acc[i][j], /*reuse_a=*/false, /*reuse_b=*/false);

    // Next buffer fully resident (this wave), then sync the workgroup.
    wait_async_copies();
    __syncthreads();
  }

  // Writeback: C/D layout -> VGPR r holds M = r (lanes 0-15) or 8+r (16-31),
  // N = lane%16. Add bias at store.
#pragma unroll
  for (int j = 0; j < 4; ++j) {
    const int cn = n0 + wn0 + j * 16 + lr;
    const float bv = bias[cn];
#pragma unroll
    for (int i = 0; i < 2; ++i) {
#pragma unroll
      for (int r = 0; r < 8; ++r) {
        const int cm = m0 + wm0 + i * 16 + lh * 8 + r;
        C[(size_t)cm * OUTF + cn] = acc[i][j][r] + bv;
      }
    }
  }
}

// ---------------- Host launcher ---------------------------------------------
extern "C" void kernel_launch(void* const* d_in, const int* in_sizes, int n_in,
                              void* d_out, int out_size, void* d_ws, size_t ws_size,
                              hipStream_t stream) {
  const float* x    = (const float*)d_in[0];   // [2048,4096]
  const float* seed = (const float*)d_in[1];   // [4096,4096]
  const float* da   = (const float*)d_in[2];   // [20]
  const float* db   = (const float*)d_in[3];   // [20]
  const float* bias = (const float*)d_in[4];   // [4096]
  float* out = (float*)d_out;                  // [2048,4096]

  // Workspace: [0,32MB) bf16 weight, [32MB,48MB) bf16 x. Fully rewritten each call.
  __bf16* wbf = (__bf16*)d_ws;
  __bf16* xbf = (__bf16*)((char*)d_ws + (size_t)OUTF * INF * sizeof(__bf16));

  fractal_transcribe_bf16<<<(OUTF * INF) / (256 * 4), 256, 0, stream>>>(seed, da, db, wbf);
  cvt_f32_to_bf16<<<(TOKENS * INF) / (256 * 4), 256, 0, stream>>>(x, xbf);
  fractal_gemm_wmma<<<dim3(OUTF / 128, TOKENS / 128), 256, 0, stream>>>(xbf, wbf, bias, out);
}
